// Model_37641093382530
// MI455X (gfx1250) — matrix operand order repaired and checked by reading, not compiled, not run on hardware
//
#include <hip/hip_runtime.h>
#include <math.h>
#include <stdint.h>

// Problem constants (match reference)
#define B_   128
#define T_   256
#define D_   128
#define H_   512
#define G4H  2048          // 4*H
#define NWG  64            // persistent grid blocks (Phase C mapping assumes exactly 64)
#define TPB  256           // 8 waves of 32
#define NWAVES (NWG * (TPB / 32))

typedef __attribute__((ext_vector_type(16))) __bf16 v16bf;
typedef __attribute__((ext_vector_type(8)))  __bf16 v8bf;
typedef __attribute__((ext_vector_type(8)))  float  v8f;
typedef unsigned int v4ui __attribute__((ext_vector_type(4)));
typedef int          v8i  __attribute__((ext_vector_type(8)));
typedef int          v4i  __attribute__((ext_vector_type(4)));

// ---------- helpers ----------
static __device__ __forceinline__ __bf16 f2bf(float f) {
  union { float f; unsigned u; } in; in.f = f;
  unsigned short s = (unsigned short)((in.u + 0x7FFFu + ((in.u >> 16) & 1u)) >> 16);
  union { unsigned short s; __bf16 b; } out; out.s = s;
  return out.b;
}

static __device__ __forceinline__ float sigm(float x) { return 1.0f / (1.0f + __expf(-x)); }

// A-fragment (16x32 bf16, row-major source [rows, ld]); p0 = row base + kbase.
// Per ISA 7.12.2: lane l holds M=l&15; k = 8*(l>>4) + {0..7} and 16 + 8*(l>>4) + {0..7}.
static __device__ __forceinline__ v16bf load_a_frag(const __bf16* p0, int lane) {
  const __bf16* p = p0 + ((lane >> 4) << 3);
  union { v16bf v; v8bf h[2]; } u;
  u.h[0] = *(const v8bf*)(p);
  u.h[1] = *(const v8bf*)(p + 16);
  return u.v;
}

// B-fragment (32x16 bf16). B = W^T, W stored row-major [N, ldK]; p0 = row(n) base + kbase.
// Per ISA: lane l holds N=l&15; k = 8*(l>>4) + {0..7} and 16 + 8*(l>>4) + {0..7}.
static __device__ __forceinline__ v16bf load_b_frag(const __bf16* __restrict__ p0, int lane) {
  const __bf16* p = p0 + ((lane >> 4) << 3);
  union { v16bf v; v8bf h[2]; } u;
  u.h[0] = *(const v8bf*)(p);
  u.h[1] = *(const v8bf*)(p + 16);
  return u.v;
}

static __device__ __forceinline__ v8f wmma_bf16(v16bf a, v16bf b, v8f c) {
  return __builtin_amdgcn_wmma_f32_16x16x32_bf16(false, a, false, b, (short)0, c, false, false);
}

// ---- Tensor Data Mover: 1-D contiguous copy global -> LDS (data_size = 1 byte) ----
// D# per CDNA5 ISA ch.8: group0 = {count/lds_addr/global_addr/type}, group1 = dims.
// tile_dim1 == 0 -> 1-D tile; tensor_dim0 == tile_dim0 == nbytes (<= 65535, mult of 4).
// This toolchain exposes the 6-arg builtin: (v4u, v8i, v4i, v4i, v8i, i32 cpol).
static __device__ __forceinline__ void tdm_load_1d(unsigned lds_off, const void* gptr,
                                                   unsigned nbytes) {
  unsigned long long ga = (unsigned long long)(uintptr_t)gptr;
  v4ui g0;
  g0[0] = 1u;                                               // count=1 (valid), user mode
  g0[1] = lds_off;                                          // lds_addr (bytes)
  g0[2] = (unsigned)(ga & 0xFFFFFFFFu);                     // global_addr[31:0]
  g0[3] = (unsigned)((ga >> 32) & 0x01FFFFFFu) | (2u << 30);// global_addr[56:32] | type=2
  v8i g1;
  g1[0] = 0;                                                // wg_mask=0, data_size=1B, no flags
  g1[1] = (int)((nbytes & 0xFFFFu) << 16);                  // tensor_dim0[15:0] @ w1[31:16]
  g1[2] = (int)((nbytes >> 16) & 0xFFFFu);                  // tensor_dim0[31:16]; tensor_dim1=0
  g1[3] = (int)((nbytes & 0xFFFFu) << 16);                  // tile_dim0 @ w3[31:16]
  g1[4] = 0;                                                // tile_dim1=0 (1-D), tile_dim2=0
  g1[5] = (int)nbytes;                                      // tensor_dim0_stride low
  g1[6] = 0;
  g1[7] = 0;
  v4i z4 = {0, 0, 0, 0};
  v8i z8 = {0, 0, 0, 0, 0, 0, 0, 0};
  __builtin_amdgcn_tensor_load_to_lds(g0, g1, z4, z4, z8, 0);
}

static __device__ __forceinline__ unsigned lds_off_of(const void* p) {
  return (unsigned)(uintptr_t)p;                            // flat LDS addr low bits = offset
}

// Monotonic-epoch grid barrier (all NWG blocks resident; counters zeroed by init kernel).
static __device__ __forceinline__ void gbar(unsigned* cnt, unsigned* flag, unsigned& epoch) {
  __syncthreads();
  if (threadIdx.x == 0) {
    __threadfence();
    unsigned target = epoch + 1;
    unsigned prev = __hip_atomic_fetch_add(cnt, 1u, __ATOMIC_ACQ_REL, __HIP_MEMORY_SCOPE_AGENT);
    if (prev == NWG - 1) {
      __hip_atomic_store(cnt, 0u, __ATOMIC_RELAXED, __HIP_MEMORY_SCOPE_AGENT);
      __hip_atomic_fetch_add(flag, 1u, __ATOMIC_ACQ_REL, __HIP_MEMORY_SCOPE_AGENT);
    } else {
      while (__hip_atomic_load(flag, __ATOMIC_ACQUIRE, __HIP_MEMORY_SCOPE_AGENT) < target)
        __builtin_amdgcn_s_sleep(2);
    }
  }
  epoch += 1;
  __syncthreads();
}

// ---------- prep kernels ----------
__global__ void k_zero(unsigned* __restrict__ p, int n) {
  int i = blockIdx.x * blockDim.x + threadIdx.x;
  if (i < n) p[i] = 0u;
}

__global__ void k_cvt(const float* __restrict__ src, __bf16* __restrict__ dst, int n) {
  int i = blockIdx.x * blockDim.x + threadIdx.x;
  if (i < n) dst[i] = f2bf(src[i]);
}

__global__ void k_cvt_offdiag(const float* __restrict__ src, __bf16* __restrict__ dst, int dim) {
  int i = blockIdx.x * blockDim.x + threadIdx.x;
  if (i < dim * dim) {
    int r = i / dim, c = i - r * dim;
    dst[i] = f2bf((r == c) ? 0.0f : src[i]);
  }
}

// Delta recurrence (scan over D per (b,t)) + per-step mask sums.
__global__ void k_delta(const float* __restrict__ data, __bf16* __restrict__ deltas,
                        float* __restrict__ masksum) {
  int gid = blockIdx.x * blockDim.x + threadIdx.x;
  if (gid >= B_ * T_) return;
  int b = gid / T_, t = gid - b * T_;
  const float* row  = data + ((size_t)b * T_ + t) * D_;
  const float* rowp = row - D_;                    // valid only if t>0
  __bf16* drow = deltas + ((size_t)b * T_ + t) * D_;
  float msum = 0.0f, carry = 0.0f;
  float xl = row[D_ - 1];
  float m_dm1 = (xl == xl) ? 1.0f : 0.0f;          // wrapped left neighbor (roll semantics)
  for (int d = 0; d < D_; ++d) {
    float x = row[d];
    float mcur = (x == x) ? 1.0f : 0.0f;
    msum += mcur;
    float v = 0.0f;
    if (t > 0) {
      float xp = rowp[d];
      float mprev = (xp == xp) ? 1.0f : 0.0f;
      v = (mprev == 0.0f) ? (carry + 1.0f) : ((m_dm1 == 1.0f) ? 1.0f : 0.0f);
    }
    carry = v;
    m_dm1 = mcur;
    drow[d] = f2bf(v);
  }
  atomicAdd(&masksum[t], msum);
}

// ---------- persistent kernel ----------
struct Ptrs {
  const float* data;
  const float* b_ih; const float* b_hh; const float* bdh; const float* bh; const float* bfv;
  const __bf16* Wih; const __bf16* Whh; const __bf16* Wdh; const __bf16* Wh; const __bf16* Wf;
  const __bf16* deltas;
  float* h; float* c; __bf16* hdec; __bf16* ccm;
  float* gates; float* masksum; float* loss;
  unsigned* barcnt; unsigned* barflag;
  float* out;
};

__global__ __launch_bounds__(TPB, 1) void brits_persistent(Ptrs P) {
  const int tid  = threadIdx.x;
  const int lane = tid & 31;
  const int widb = tid >> 5;
  const int gwid = blockIdx.x * (TPB / 32) + widb;
  unsigned epoch = 0;

  // LDS staging: Phase B uses [hdec 16xH | xc 16xD]; Phase C uses [ccm 16x2D | hdec 16xH]
  __shared__ __align__(32) __bf16 smem[16 * (2 * D_) + 16 * H_];   // 24 KB

#pragma unroll 1
  for (int t = 0; t < T_; ++t) {
    // ---- Phase A: gamma_h = exp(-relu(d @ Wdh^T + bdh)); h_dec = h * gamma_h (bf16) ----
    for (int tile = gwid; tile < (B_ / 16) * (H_ / 16); tile += NWAVES) {
      int bt = tile >> 5, nt = tile & 31;
      v8f acc = {};
      const __bf16* arow = P.deltas + ((size_t)(bt * 16 + (lane & 15)) * T_ + t) * D_;
      const __bf16* brow = P.Wdh + (size_t)(nt * 16 + (lane & 15)) * D_;
#pragma unroll
      for (int kb = 0; kb < D_; kb += 32)
        acc = wmma_bf16(load_a_frag(arow + kb, lane), load_b_frag(brow + kb, lane), acc);
      int n = nt * 16 + (lane & 15);
      float bias = P.bdh[n];
#pragma unroll
      for (int r = 0; r < 8; ++r) {
        int b = bt * 16 + r + ((lane >> 4) << 3);
        float g  = __expf(-fmaxf(acc[r] + bias, 0.0f));
        float hd = P.h[b * H_ + n] * g;
        P.hdec[b * H_ + n] = f2bf(hd);
      }
    }
    gbar(P.barcnt, P.barflag, epoch);

    // ---- Phase B (blocks 0..7): x_h -> loss1 -> x_c (LDS) -> z_h -> loss2 -> c_c ----
    if (blockIdx.x < (B_ / 16)) {
      const int bt = blockIdx.x;
      const int w  = widb;                     // col tile 0..7 of D
      const int n  = w * 16 + (lane & 15);
      __bf16* s_hdec = smem;                   // 16 x H_
      __bf16* s_xc   = smem + 16 * H_;         // 16 x D_

      // TDM: stage the 16-row h_dec activation tile (16 KB) into LDS once per block.
      if (widb == 0) {
        tdm_load_1d(lds_off_of(s_hdec), P.hdec + (size_t)bt * 16 * H_, 16 * H_ * 2);
        __builtin_amdgcn_s_wait_tensorcnt(0);
      }
      __syncthreads();

      float inv = 1.0f / (P.masksum[t] + 1e-5f);

      // x_h = sigmoid(h_dec @ Wh^T + bh), K = H  (A from LDS, B from L2)
      v8f acc = {};
      {
        const __bf16* arow = s_hdec + (size_t)(lane & 15) * H_;
        const __bf16* brow = P.Wh + (size_t)n * H_;
#pragma unroll 4
        for (int kb = 0; kb < H_; kb += 32)
          acc = wmma_bf16(load_a_frag(arow + kb, lane), load_b_frag(brow + kb, lane), acc);
      }
      float xhv[8], xnv[8], mv[8];
      float l1 = 0.0f;
#pragma unroll
      for (int r = 0; r < 8; ++r) {
        int br = r + ((lane >> 4) << 3);             // local batch row 0..15
        int b  = bt * 16 + br;
        xhv[r] = sigm(acc[r] + P.bh[n]);
        float x = P.data[((size_t)b * T_ + t) * D_ + n];
        mv[r]  = (x == x) ? 1.0f : 0.0f;
        xnv[r] = (x == x) ? x : 0.0f;
        l1 += fabsf(xnv[r] - xhv[r]) * mv[r];
        float xcv = (mv[r] > 0.0f) ? xnv[r] : xhv[r];
        s_xc[br * D_ + n] = f2bf(xcv);
      }
#pragma unroll
      for (int off = 16; off > 0; off >>= 1) l1 += __shfl_xor(l1, off, 32);
      if (lane == 0) atomicAdd(P.loss, l1 * inv);
      __syncthreads();   // x_c tile complete in LDS

      // z_h = sigmoid(x_c @ Wf_m^T + bf), K = D  (A from LDS)
      v8f accz = {};
      {
        const __bf16* arow = s_xc + (size_t)(lane & 15) * D_;
        const __bf16* brow = P.Wf + (size_t)n * D_;
#pragma unroll
        for (int kb = 0; kb < D_; kb += 32)
          accz = wmma_bf16(load_a_frag(arow + kb, lane), load_b_frag(brow + kb, lane), accz);
      }
      float l2 = 0.0f;
#pragma unroll
      for (int r = 0; r < 8; ++r) {
        int b = bt * 16 + r + ((lane >> 4) << 3);
        float z  = sigm(accz[r] + P.bfv[n]);
        float ch = fabsf(0.5f * (z + xhv[r]));
        float cc = (mv[r] > 0.0f) ? xnv[r] : ch;
        P.out[1 + ((size_t)b * T_ + t) * D_ + n] = cc;    // imputations
        l2 += fabsf(xnv[r] - z) * mv[r];
        P.ccm[b * (2 * D_) + n]      = f2bf(cc);          // gates input = [c_c, m]
        P.ccm[b * (2 * D_) + D_ + n] = f2bf(mv[r]);
      }
#pragma unroll
      for (int off = 16; off > 0; off >>= 1) l2 += __shfl_xor(l2, off, 32);
      if (lane == 0) atomicAdd(P.loss, l2 * inv);
    }
    gbar(P.barcnt, P.barflag, epoch);

    // ---- Phase C: gates = [c_c,m] @ Wih^T + h_dec @ Whh^T + b_ih + b_hh ----
    // Block-uniform batch tile: bt = blk&7, nt span = (blk>>3)*16 .. +15 (NWG==64).
    {
      const int bt  = blockIdx.x & 7;
      const int nt0 = (blockIdx.x >> 3) << 4;
      __bf16* s_ccm  = smem;                   // 16 x 2D
      __bf16* s_hdec = smem + 16 * (2 * D_);   // 16 x H

      if (widb == 0) {
        tdm_load_1d(lds_off_of(s_ccm),  P.ccm  + (size_t)bt * 16 * (2 * D_), 16 * (2 * D_) * 2);
        tdm_load_1d(lds_off_of(s_hdec), P.hdec + (size_t)bt * 16 * H_,       16 * H_ * 2);
        __builtin_amdgcn_s_wait_tensorcnt(0);
      }
      __syncthreads();

#pragma unroll 1
      for (int sub = 0; sub < 2; ++sub) {
        const int nt = nt0 + widb + (sub << 3);
        v8f acc = {};
        {
          const __bf16* arow = s_ccm + (size_t)(lane & 15) * (2 * D_);
          const __bf16* brow = P.Wih + (size_t)(nt * 16 + (lane & 15)) * (2 * D_);
#pragma unroll
          for (int kb = 0; kb < 2 * D_; kb += 32)
            acc = wmma_bf16(load_a_frag(arow + kb, lane), load_b_frag(brow + kb, lane), acc);
        }
        {
          const __bf16* arow = s_hdec + (size_t)(lane & 15) * H_;
          const __bf16* brow = P.Whh + (size_t)(nt * 16 + (lane & 15)) * H_;
#pragma unroll 4
          for (int kb = 0; kb < H_; kb += 32)
            acc = wmma_bf16(load_a_frag(arow + kb, lane), load_b_frag(brow + kb, lane), acc);
        }
        int n = nt * 16 + (lane & 15);
        float bias = P.b_ih[n] + P.b_hh[n];
#pragma unroll
        for (int r = 0; r < 8; ++r) {
          int b = bt * 16 + r + ((lane >> 4) << 3);
          P.gates[b * G4H + n] = acc[r] + bias;
        }
      }
    }
    gbar(P.barcnt, P.barflag, epoch);

    // ---- Phase D: LSTM cell elementwise ----
    for (int e = blockIdx.x * TPB + tid; e < B_ * H_; e += NWG * TPB) {
      int b = e >> 9, n = e & (H_ - 1);
      const float* gr = P.gates + (size_t)b * G4H;
      float ig = sigm(gr[n]);
      float fg = sigm(gr[H_ + n]);
      float gg = tanhf(gr[2 * H_ + n]);
      float og = sigm(gr[3 * H_ + n]);
      float cn = fg * P.c[e] + ig * gg;
      P.c[e] = cn;
      P.h[e] = og * tanhf(cn);
    }
    gbar(P.barcnt, P.barflag, epoch);
  }

  if (blockIdx.x == 0 && tid == 0) P.out[0] = (*P.loss) * 0.3f;   // IMPUTED_WEIGHT
}

// ---------- host ----------
extern "C" void kernel_launch(void* const* d_in, const int* in_sizes, int n_in,
                              void* d_out, int out_size, void* d_ws, size_t ws_size,
                              hipStream_t stream) {
  const float* data = (const float*)d_in[0];
  const float* W_ih = (const float*)d_in[1];
  const float* b_ih = (const float*)d_in[2];
  const float* W_hh = (const float*)d_in[3];
  const float* b_hh = (const float*)d_in[4];
  const float* Wdh  = (const float*)d_in[5];
  const float* bdh  = (const float*)d_in[6];
  // d_in[7] Wdx, d_in[8] bdx : dead code (feed only unused _alpha / gamma_x)
  const float* Wh   = (const float*)d_in[9];
  const float* bh   = (const float*)d_in[10];
  const float* Wf   = (const float*)d_in[11];
  const float* bf   = (const float*)d_in[12];
  // d_in[13] Wc, d_in[14] bc : dead code

  char* ws = (char*)d_ws;
  size_t off = 0;
  auto alloc = [&](size_t bytes) { size_t cur = off; off = (off + bytes + 255) & ~(size_t)255; return cur; };

  size_t o_bar   = alloc(2 * sizeof(unsigned));
  size_t o_loss  = alloc(sizeof(float));
  size_t o_msum  = alloc(T_ * sizeof(float));
  size_t o_h     = alloc((size_t)B_ * H_ * 4);
  size_t o_c     = alloc((size_t)B_ * H_ * 4);
  size_t zero_words = (o_c + (size_t)B_ * H_ * 4) / 4;   // bar..c contiguous prefix
  size_t o_hdec  = alloc((size_t)B_ * H_ * 2);
  size_t o_ccm   = alloc((size_t)B_ * 2 * D_ * 2);
  size_t o_gates = alloc((size_t)B_ * G4H * 4);
  size_t o_delt  = alloc((size_t)B_ * T_ * D_ * 2);
  size_t o_Wih   = alloc((size_t)G4H * 2 * D_ * 2);
  size_t o_Whh   = alloc((size_t)G4H * H_ * 2);
  size_t o_Wdh   = alloc((size_t)H_ * D_ * 2);
  size_t o_Wh    = alloc((size_t)D_ * H_ * 2);
  size_t o_Wf    = alloc((size_t)D_ * D_ * 2);
  (void)ws_size; (void)out_size; (void)n_in; (void)in_sizes;

  // 1) zero barriers / loss / masksum / h0 / c0
  k_zero<<<(int)((zero_words + 255) / 256), 256, 0, stream>>>((unsigned*)ws, (int)zero_words);

  // 2) weight conversions to bf16 (Wf with off-diagonal mask)
  {
    int n;
    n = G4H * 2 * D_; k_cvt<<<(n + 255) / 256, 256, 0, stream>>>(W_ih, (__bf16*)(ws + o_Wih), n);
    n = G4H * H_;     k_cvt<<<(n + 255) / 256, 256, 0, stream>>>(W_hh, (__bf16*)(ws + o_Whh), n);
    n = H_ * D_;      k_cvt<<<(n + 255) / 256, 256, 0, stream>>>(Wdh,  (__bf16*)(ws + o_Wdh), n);
    n = D_ * H_;      k_cvt<<<(n + 255) / 256, 256, 0, stream>>>(Wh,   (__bf16*)(ws + o_Wh),  n);
    n = D_ * D_;      k_cvt_offdiag<<<(n + 255) / 256, 256, 0, stream>>>(Wf, (__bf16*)(ws + o_Wf), D_);
  }

  // 3) delta scan + per-step mask sums
  k_delta<<<(B_ * T_ + 255) / 256, 256, 0, stream>>>(data, (__bf16*)(ws + o_delt),
                                                     (float*)(ws + o_msum));

  // 4) persistent recurrence
  Ptrs P;
  P.data = data;
  P.b_ih = b_ih; P.b_hh = b_hh; P.bdh = bdh; P.bh = bh; P.bfv = bf;
  P.Wih = (const __bf16*)(ws + o_Wih); P.Whh = (const __bf16*)(ws + o_Whh);
  P.Wdh = (const __bf16*)(ws + o_Wdh); P.Wh  = (const __bf16*)(ws + o_Wh);
  P.Wf  = (const __bf16*)(ws + o_Wf);
  P.deltas = (const __bf16*)(ws + o_delt);
  P.h = (float*)(ws + o_h); P.c = (float*)(ws + o_c);
  P.hdec = (__bf16*)(ws + o_hdec); P.ccm = (__bf16*)(ws + o_ccm);
  P.gates = (float*)(ws + o_gates); P.masksum = (float*)(ws + o_msum);
  P.loss = (float*)(ws + o_loss);
  P.barcnt = (unsigned*)(ws + o_bar); P.barflag = ((unsigned*)(ws + o_bar)) + 1;
  P.out = (float*)d_out;

  brits_persistent<<<NWG, TPB, 0, stream>>>(P);
}
